// ExpressionAttentionLayer_5531917877944
// MI455X (gfx1250) — matrix operand order repaired and checked by reading, not compiled, not run on hardware
//
#include <hip/hip_runtime.h>
#include <hip/hip_bf16.h>

#define BATCH 2
#define SEQ   2048
#define DIM   512
#define HEADS 8
#define HDIM  64
#define EPSL  1e-8f

typedef __bf16 bf16_t;
typedef __attribute__((ext_vector_type(16))) __bf16 v16bf;
typedef __attribute__((ext_vector_type(8)))  __bf16 v8bf;
typedef __attribute__((ext_vector_type(8)))  float  v8f;

union AFrag { v16bf v; v8bf h[2]; };

// Native bf16 convert (RNE fptrunc -> v_cvt_pk_bf16_f32 on gfx1250)
__device__ __forceinline__ bf16_t f2bf(float f) { return (bf16_t)f; }

__device__ __forceinline__ v8bf cvt8(v8f f) {
  v8bf o;
#pragma unroll
  for (int i = 0; i < 8; ++i) o[i] = (bf16_t)f[i];
  return o;
}

__device__ __forceinline__ v8f vzero8() {
  v8f c;
#pragma unroll
  for (int i = 0; i < 8; ++i) c[i] = 0.0f;
  return c;
}

// ---------------------------------------------------------------------------
// Prep: transpose fp32 weight [K][N] -> bf16 [N][K] (so WMMA B gathers are
// one contiguous 32B load per lane).
// ---------------------------------------------------------------------------
__global__ void k_transpose_w(const float* __restrict__ W, bf16_t* __restrict__ Wt,
                              int K, int N) {
  int idx = blockIdx.x * blockDim.x + threadIdx.x;
  if (idx >= K * N) return;
  int k = idx % K;
  int n = idx / K;
  Wt[idx] = f2bf(W[(size_t)k * N + n]);          // Wt[n*K + k]
}

// Prep: V [B,S,H,HD] fp32 -> Vt [B,H,HD,S] bf16 (K=key contiguous for AV WMMA)
__global__ void k_transpose_v(const float* __restrict__ V, bf16_t* __restrict__ Vt) {
  int idx = blockIdx.x * blockDim.x + threadIdx.x;
  if (idx >= BATCH * HEADS * HDIM * SEQ) return;
  int s  = idx & (SEQ - 1);
  int d  = (idx >> 11) & (HDIM - 1);
  int bh = idx >> 17;
  int b = bh >> 3, h = bh & 7;
  Vt[idx] = f2bf(V[(((size_t)(b * SEQ + s)) * HEADS + h) * HDIM + d]);
}

// ---------------------------------------------------------------------------
// Register-blocked, software-pipelined WMMA tile loop: one wave computes a
// 16x64 output strip. A bf16 row-major [M][lda]; Bt bf16 N-major [N][ldb]
// (ldb == K). Loads for K-step k0 are issued while WMMAs for k0-32 execute.
// ---------------------------------------------------------------------------
__device__ __forceinline__ void gemm_tile4(const bf16_t* __restrict__ A, int lda,
                                           const bf16_t* __restrict__ Bt, int ldb,
                                           int m0, int n0, int K, int lane,
                                           v8f c[4]) {
  const int r  = lane & 15;
  const int hi = lane >> 4;
#pragma unroll
  for (int nt = 0; nt < 4; ++nt) c[nt] = vzero8();
  const bf16_t* aRow = A + (size_t)(m0 + r) * lda + hi * 16;
  const bf16_t* bRow = Bt + (size_t)(n0 + r) * ldb + hi * 16;

  AFrag a;
  v16bf bf[4];
  a.h[0] = *(const v8bf*)(aRow);               // K = kb .. kb+7
  a.h[1] = *(const v8bf*)(aRow + 8);           // K = kb+8 .. kb+15
#pragma unroll
  for (int nt = 0; nt < 4; ++nt)
    bf[nt] = *(const v16bf*)(bRow + (size_t)nt * 16 * ldb);

  for (int k0 = 32; k0 < K; k0 += 32) {
    AFrag an;
    v16bf bn[4];
    an.h[0] = *(const v8bf*)(aRow + k0);
    an.h[1] = *(const v8bf*)(aRow + k0 + 8);
#pragma unroll
    for (int nt = 0; nt < 4; ++nt)
      bn[nt] = *(const v16bf*)(bRow + (size_t)nt * 16 * ldb + k0);
#pragma unroll
    for (int nt = 0; nt < 4; ++nt)
      c[nt] = __builtin_amdgcn_wmma_f32_16x16x32_bf16(false, a.v, false, bf[nt],
                                                      (short)0, c[nt], false, false);
    a = an;
#pragma unroll
    for (int nt = 0; nt < 4; ++nt) bf[nt] = bn[nt];
  }
#pragma unroll
  for (int nt = 0; nt < 4; ++nt)
    c[nt] = __builtin_amdgcn_wmma_f32_16x16x32_bf16(false, a.v, false, bf[nt],
                                                    (short)0, c[nt], false, false);
}

// ---------------------------------------------------------------------------
// Kernel 1: fused = concat(gene,expr) @ fused_W + b   -> bf16 [4096][512]
// A is fp32 in global (two sources), converted on the fly; same one-deep
// software pipeline as gemm_tile4.
// ---------------------------------------------------------------------------
__global__ void __launch_bounds__(128)
k_fused_gemm(const float* __restrict__ gene, const float* __restrict__ expr,
             const bf16_t* __restrict__ Wt, const float* __restrict__ bias,
             bf16_t* __restrict__ out) {
  int wave = (blockIdx.x * blockDim.x + threadIdx.x) >> 5;   // [0, 2048)
  int lane = threadIdx.x & 31;
  int tn = wave & 7, tm = wave >> 3;
  int m0 = tm * 16, n0 = tn * 64;
  int r = lane & 15, hi = lane >> 4;
  int mrow = m0 + r;
  int b = mrow >> 11, s = mrow & (SEQ - 1);
  const float* g = gene + ((size_t)(b * SEQ + s)) * DIM + hi * 16;
  const float* e = expr + ((size_t)(b * SEQ + s)) * DIM + hi * 16;
  const bf16_t* bRow = Wt + (size_t)(n0 + r) * (2 * DIM) + hi * 16;

  v8f c[4];
#pragma unroll
  for (int nt = 0; nt < 4; ++nt) c[nt] = vzero8();

  AFrag a;
  v16bf bf[4];
  {
    v8f f0 = *(const v8f*)(g);
    v8f f1 = *(const v8f*)(g + 8);
    a.h[0] = cvt8(f0);
    a.h[1] = cvt8(f1);
#pragma unroll
    for (int nt = 0; nt < 4; ++nt)
      bf[nt] = *(const v16bf*)(bRow + (size_t)nt * 16 * (2 * DIM));
  }
  for (int k0 = 32; k0 < 2 * DIM; k0 += 32) {
    const float* src = (k0 < DIM) ? (g + k0) : (e + (k0 - DIM));
    v8f f0 = *(const v8f*)(src);
    v8f f1 = *(const v8f*)(src + 8);
    AFrag an;
    an.h[0] = cvt8(f0);
    an.h[1] = cvt8(f1);
    v16bf bn[4];
#pragma unroll
    for (int nt = 0; nt < 4; ++nt)
      bn[nt] = *(const v16bf*)(bRow + (size_t)nt * 16 * (2 * DIM) + k0);
#pragma unroll
    for (int nt = 0; nt < 4; ++nt)
      c[nt] = __builtin_amdgcn_wmma_f32_16x16x32_bf16(false, a.v, false, bf[nt],
                                                      (short)0, c[nt], false, false);
    a = an;
#pragma unroll
    for (int nt = 0; nt < 4; ++nt) bf[nt] = bn[nt];
  }
#pragma unroll
  for (int nt = 0; nt < 4; ++nt)
    c[nt] = __builtin_amdgcn_wmma_f32_16x16x32_bf16(false, a.v, false, bf[nt],
                                                    (short)0, c[nt], false, false);

#pragma unroll
  for (int nt = 0; nt < 4; ++nt) {
    float bn = bias[n0 + nt * 16 + r];
#pragma unroll
    for (int rr = 0; rr < 8; ++rr) {
      int row = m0 + (hi ? 8 + rr : rr);
      out[(size_t)row * DIM + n0 + nt * 16 + r] = f2bf(c[nt][rr] + bn);
    }
  }
}

// ---------------------------------------------------------------------------
// Kernel 2: q/k projections. blockIdx.y: 0 = q (scaled 1/sqrt(HD)), 1 = k.
// Output bf16 laid out [B,H,S,HD] for the attention kernel.
// ---------------------------------------------------------------------------
__global__ void __launch_bounds__(128)
k_qk_gemm(const bf16_t* __restrict__ fusedb,
          const bf16_t* __restrict__ Wqt, const bf16_t* __restrict__ Wkt,
          const float* __restrict__ bq, const float* __restrict__ bk,
          bf16_t* __restrict__ qout, bf16_t* __restrict__ kout) {
  int wave = (blockIdx.x * blockDim.x + threadIdx.x) >> 5;   // [0, 2048)
  int lane = threadIdx.x & 31;
  int sel = blockIdx.y;
  int tn = wave & 7, tm = wave >> 3;
  int m0 = tm * 16, n0 = tn * 64;
  int r = lane & 15, hi = lane >> 4;

  const bf16_t* Wt = sel ? Wkt : Wqt;
  const float* bias = sel ? bk : bq;
  bf16_t* dst = sel ? kout : qout;
  float scale = sel ? 1.0f : 0.125f;          // 1/sqrt(64)

  v8f c[4];
  gemm_tile4(fusedb, DIM, Wt, DIM, m0, n0, DIM, lane, c);

#pragma unroll
  for (int nt = 0; nt < 4; ++nt) {
    int n = n0 + nt * 16 + r;
    int h = n >> 6, hd = n & (HDIM - 1);
    float bn = bias[n];
#pragma unroll
    for (int rr = 0; rr < 8; ++rr) {
      int row = m0 + (hi ? 8 + rr : rr);
      int b = row >> 11, s = row & (SEQ - 1);
      dst[(((size_t)(b * HEADS + h)) * SEQ + s) * HDIM + hd] = f2bf((c[nt][rr] + bn) * scale);
    }
  }
}

// ---------------------------------------------------------------------------
// Kernel 3: flash attention with mask + L1 renorm fused.
// One wave per 16-row q tile. Per 32-key block: 4 QK^T WMMAs, online
// softmax stats via shfl_xor, P=exp*M staged bf16 in LDS (C->A layout swap),
// 4 AV WMMAs. V-tile loads are hoisted to the top of the iteration so the
// softmax/shfl/LDS work covers their latency. Final:
// O / (sum(exp*M) + eps*sum(exp)).
// ---------------------------------------------------------------------------
__global__ void __launch_bounds__(128)
k_attn(const bf16_t* __restrict__ qb, const bf16_t* __restrict__ kbm,
       const bf16_t* __restrict__ vt, const float* __restrict__ Mm,
       bf16_t* __restrict__ attnb) {
  __shared__ alignas(32) bf16_t Plds[4][16 * 32];
  int wave = threadIdx.x >> 5;
  int lane = threadIdx.x & 31;
  int t = blockIdx.x * 4 + wave;              // [0, 2048)
  int qt = t & 127;                           // S/16 tiles
  int bh = t >> 7;                            // [0, 16)
  int b = bh >> 3, h = bh & 7;
  int r = lane & 15, hi = lane >> 4;

  // q A-fragments (reused for all key blocks)
  const bf16_t* qrow = qb + ((size_t)bh * SEQ + qt * 16 + r) * HDIM + hi * 16;
  AFrag aq0, aq1;
  aq0.h[0] = *(const v8bf*)(qrow);
  aq0.h[1] = *(const v8bf*)(qrow + 8);
  aq1.h[0] = *(const v8bf*)(qrow + 32);
  aq1.h[1] = *(const v8bf*)(qrow + 40);

  const bf16_t* kbase = kbm + (size_t)bh * SEQ * HDIM;
  const bf16_t* vbase = vt + (size_t)bh * HDIM * SEQ;
  const float*  Mbase = Mm + ((size_t)b * SEQ + qt * 16) * SEQ;

  float mrun[8], lsum[8], lmsum[8];
#pragma unroll
  for (int i = 0; i < 8; ++i) { mrun[i] = -1e30f; lsum[i] = 0.f; lmsum[i] = 0.f; }
  v8f o[4];
#pragma unroll
  for (int i = 0; i < 4; ++i) o[i] = vzero8();
  bf16_t* P = &Plds[wave][0];

  for (int kb32 = 0; kb32 < SEQ; kb32 += 32) {
    // prefetch next key-block's mask rows (read-once stream)
    if (kb32 + 32 < SEQ)
      __builtin_prefetch(Mbase + (size_t)(hi * 8) * SEQ + kb32 + 32 + r, 0, 0);

    // --- issue all global loads for this block up front ---
    const bf16_t* kp0 = kbase + (size_t)(kb32 + r) * HDIM + hi * 16;
    const bf16_t* kp1 = kbase + (size_t)(kb32 + 16 + r) * HDIM + hi * 16;
    v16bf kb0a = *(const v16bf*)(kp0);
    v16bf kb0b = *(const v16bf*)(kp0 + 32);
    v16bf kb1a = *(const v16bf*)(kp1);
    v16bf kb1b = *(const v16bf*)(kp1 + 32);
    v16bf bv[4];
#pragma unroll
    for (int nt = 0; nt < 4; ++nt)
      bv[nt] = *(const v16bf*)(vbase + (size_t)(nt * 16 + r) * SEQ + kb32 + hi * 16);
    float Mv0[8], Mv1[8];
#pragma unroll
    for (int rr = 0; rr < 8; ++rr) {
      int row = hi ? 8 + rr : rr;
      Mv0[rr] = __builtin_nontemporal_load(Mbase + (size_t)row * SEQ + kb32 + r);
      Mv1[rr] = __builtin_nontemporal_load(Mbase + (size_t)row * SEQ + kb32 + 16 + r);
    }

    // --- scores for two 16-key tiles (K = HDIM = 64 -> 2 wmma each) ---
    v8f s0 = vzero8(), s1 = vzero8();
    s0 = __builtin_amdgcn_wmma_f32_16x16x32_bf16(false, aq0.v, false, kb0a, (short)0, s0, false, false);
    s0 = __builtin_amdgcn_wmma_f32_16x16x32_bf16(false, aq1.v, false, kb0b, (short)0, s0, false, false);
    s1 = __builtin_amdgcn_wmma_f32_16x16x32_bf16(false, aq0.v, false, kb1a, (short)0, s1, false, false);
    s1 = __builtin_amdgcn_wmma_f32_16x16x32_bf16(false, aq1.v, false, kb1b, (short)0, s1, false, false);

    // --- online softmax stats (row reductions within 16-lane halves) ---
    float tmax[8];
#pragma unroll
    for (int rr = 0; rr < 8; ++rr) tmax[rr] = fmaxf(s0[rr], s1[rr]);
#pragma unroll
    for (int msk = 1; msk < 16; msk <<= 1)
#pragma unroll
      for (int rr = 0; rr < 8; ++rr)
        tmax[rr] = fmaxf(tmax[rr], __shfl_xor(tmax[rr], msk, 32));

    float fac[8], p0[8], p1[8], ps[8], pm[8];
#pragma unroll
    for (int rr = 0; rr < 8; ++rr) {
      float nm = fmaxf(mrun[rr], tmax[rr]);
      fac[rr] = __expf(mrun[rr] - nm);
      mrun[rr] = nm;
      p0[rr] = __expf(s0[rr] - nm);
      p1[rr] = __expf(s1[rr] - nm);
      ps[rr] = p0[rr] + p1[rr];
      pm[rr] = p0[rr] * Mv0[rr] + p1[rr] * Mv1[rr];
    }
#pragma unroll
    for (int msk = 1; msk < 16; msk <<= 1)
#pragma unroll
      for (int rr = 0; rr < 8; ++rr) {
        ps[rr] += __shfl_xor(ps[rr], msk, 32);
        pm[rr] += __shfl_xor(pm[rr], msk, 32);
      }
#pragma unroll
    for (int rr = 0; rr < 8; ++rr) {
      lsum[rr]  = lsum[rr]  * fac[rr] + ps[rr];
      lmsum[rr] = lmsum[rr] * fac[rr] + pm[rr];
    }
#pragma unroll
    for (int nt = 0; nt < 4; ++nt)
#pragma unroll
      for (int rr = 0; rr < 8; ++rr) o[nt][rr] *= fac[rr];

    // --- stage P = exp*M to LDS (C layout -> A layout swap) ---
#pragma unroll
    for (int rr = 0; rr < 8; ++rr) {
      int row = hi ? 8 + rr : rr;
      P[row * 32 + r]      = f2bf(p0[rr] * Mv0[rr]);
      P[row * 32 + 16 + r] = f2bf(p1[rr] * Mv1[rr]);
    }
    asm volatile("s_wait_dscnt 0x0" ::: "memory");

    AFrag ap;
    ap.h[0] = *(const v8bf*)(P + r * 32 + hi * 16);
    ap.h[1] = *(const v8bf*)(P + r * 32 + hi * 16 + 8);

    // --- accumulate O += P @ V (V fragments already resident) ---
#pragma unroll
    for (int nt = 0; nt < 4; ++nt)
      o[nt] = __builtin_amdgcn_wmma_f32_16x16x32_bf16(false, ap.v, false, bv[nt],
                                                      (short)0, o[nt], false, false);
  }

  // --- final normalize + store as [b, s, h*HD + d] bf16 ---
  float inv[8];
#pragma unroll
  for (int rr = 0; rr < 8; ++rr) inv[rr] = 1.0f / (lmsum[rr] + EPSL * lsum[rr]);
#pragma unroll
  for (int nt = 0; nt < 4; ++nt)
#pragma unroll
    for (int rr = 0; rr < 8; ++rr) {
      int row = hi ? 8 + rr : rr;
      size_t srow = (size_t)(b * SEQ + qt * 16 + row);
      attnb[srow * DIM + h * HDIM + nt * 16 + r] = f2bf(o[nt][rr] * inv[rr]);
    }
}

// ---------------------------------------------------------------------------
// Kernel 4: out = attn @ out_W + out_b  (fp32 output)
// ---------------------------------------------------------------------------
__global__ void __launch_bounds__(128)
k_out_gemm(const bf16_t* __restrict__ attnb, const bf16_t* __restrict__ Wot,
           const float* __restrict__ bias, float* __restrict__ out) {
  int wave = (blockIdx.x * blockDim.x + threadIdx.x) >> 5;   // [0, 2048)
  int lane = threadIdx.x & 31;
  int tn = wave & 7, tm = wave >> 3;
  int m0 = tm * 16, n0 = tn * 64;
  int r = lane & 15, hi = lane >> 4;

  v8f c[4];
  gemm_tile4(attnb, DIM, Wot, DIM, m0, n0, DIM, lane, c);

#pragma unroll
  for (int nt = 0; nt < 4; ++nt) {
    float bn = bias[n0 + nt * 16 + r];
#pragma unroll
    for (int rr = 0; rr < 8; ++rr) {
      int row = m0 + (hi ? 8 + rr : rr);
      out[(size_t)row * DIM + n0 + nt * 16 + r] = c[nt][rr] + bn;
    }
  }
}

// ---------------------------------------------------------------------------
extern "C" void kernel_launch(void* const* d_in, const int* in_sizes, int n_in,
                              void* d_out, int out_size, void* d_ws, size_t ws_size,
                              hipStream_t stream) {
  const float* gene = (const float*)d_in[0];
  const float* expr = (const float*)d_in[1];
  const float* V    = (const float*)d_in[2];
  const float* M    = (const float*)d_in[3];
  const float* fW   = (const float*)d_in[4];
  const float* fb   = (const float*)d_in[5];
  const float* Wq   = (const float*)d_in[6];
  const float* bq   = (const float*)d_in[7];
  const float* Wk   = (const float*)d_in[8];
  const float* bk   = (const float*)d_in[9];
  const float* oW   = (const float*)d_in[10];
  const float* ob   = (const float*)d_in[11];

  uint8_t* w = (uint8_t*)d_ws;
  size_t off = 0;
  auto carve = [&](size_t bytes) {
    uint8_t* p = w + off;
    off += (bytes + 255) & ~(size_t)255;
    return p;
  };
  const size_t actBytes = (size_t)BATCH * SEQ * DIM * sizeof(bf16_t);   // 4 MB
  bf16_t* fusedb = (bf16_t*)carve(actBytes);
  bf16_t* qbuf   = (bf16_t*)carve(actBytes);
  bf16_t* kbuf   = (bf16_t*)carve(actBytes);
  bf16_t* vtb    = (bf16_t*)carve(actBytes);
  bf16_t* attnb  = (bf16_t*)carve(actBytes);
  bf16_t* Wft    = (bf16_t*)carve((size_t)2 * DIM * DIM * sizeof(bf16_t));
  bf16_t* Wqt    = (bf16_t*)carve((size_t)DIM * DIM * sizeof(bf16_t));
  bf16_t* Wkt    = (bf16_t*)carve((size_t)DIM * DIM * sizeof(bf16_t));
  bf16_t* Wot    = (bf16_t*)carve((size_t)DIM * DIM * sizeof(bf16_t));

  // --- prep: transpose weights and V into bf16 WMMA-friendly layouts ---
  {
    int n1 = 2 * DIM * DIM;
    k_transpose_w<<<(n1 + 255) / 256, 256, 0, stream>>>(fW, Wft, 2 * DIM, DIM);
    int n2 = DIM * DIM;
    k_transpose_w<<<(n2 + 255) / 256, 256, 0, stream>>>(Wq, Wqt, DIM, DIM);
    k_transpose_w<<<(n2 + 255) / 256, 256, 0, stream>>>(Wk, Wkt, DIM, DIM);
    k_transpose_w<<<(n2 + 255) / 256, 256, 0, stream>>>(oW, Wot, DIM, DIM);
    int n3 = BATCH * HEADS * HDIM * SEQ;
    k_transpose_v<<<(n3 + 255) / 256, 256, 0, stream>>>(V, vtb);
  }

  // --- pipeline: fused -> q/k -> flash attention -> output projection ---
  k_fused_gemm<<<512, 128, 0, stream>>>(gene, expr, Wft, fb, fusedb);
  k_qk_gemm<<<dim3(512, 2), 128, 0, stream>>>(fusedb, Wqt, Wkt, bq, bk, qbuf, kbuf);
  k_attn<<<512, 128, 0, stream>>>(qbuf, kbuf, vtb, M, attnb);
  k_out_gemm<<<512, 128, 0, stream>>>(attnb, Wot, ob, (float*)d_out);
}
